// ImplicitMLP_48576080118009
// MI455X (gfx1250) — matrix operand order repaired and checked by reading, not compiled, not run on hardware
//
#include <hip/hip_runtime.h>

typedef _Float16 f16;
typedef __attribute__((ext_vector_type(16))) _Float16 v16h;
typedef __attribute__((ext_vector_type(8)))  _Float16 v8h;
typedef __attribute__((ext_vector_type(8)))  float    v8f;
typedef __attribute__((ext_vector_type(4)))  unsigned int u32x4;
typedef __attribute__((ext_vector_type(8)))  int i32x8;
typedef __attribute__((ext_vector_type(4)))  int i32x4;

#define TWO_PI 6.283185307179586f

#if __has_builtin(__builtin_amdgcn_tensor_load_to_lds) && \
    __has_builtin(__builtin_amdgcn_s_wait_tensorcnt)
#define USE_TDM 1
#endif

// ---------------- LDS layout (bytes) ----------------
#define LDS_F      0
#define LDS_W1     3072                        // 256 rows x 264 halves
#define LDS_W2     (LDS_W1 + 256*264*2)        // 128 rows x 264 halves
#define LDS_W3     (LDS_W2 + 128*264*2)        // 32 rows x 136 halves
#define LDS_W4     (LDS_W3 + 32*136*2)         // 16 rows x 40 halves
#define LDS_ACT    (LDS_W4 + 16*40*2)          // 8 waves x 16 x 264 halves
#define LDS_TOTAL  (LDS_ACT + 8*16*264*2)      // = 283392 bytes (< 320KB/WGP)

// float-region indices
#define F_BMAT 0
#define F_B1   256
#define F_B2   512
#define F_B3   640
#define F_B4   672
#define F_W5   688
#define F_B5   736

// d_ws half-region offsets
#define WS_W1  0
#define WS_W2  65536
#define WS_W3  98304
#define WS_W4  102400
#define WS_END 102912

// =====================================================================
// Kernel A: materialize low-rank products + f16 conversion of weights.
// =====================================================================
__global__ void prep_weights(const float* __restrict__ L1, const float* __restrict__ R1,
                             const float* __restrict__ L2, const float* __restrict__ R2,
                             const float* __restrict__ W3, const float* __restrict__ W4,
                             f16* __restrict__ wsh) {
  int i = blockIdx.x * blockDim.x + threadIdx.x;
  if (i < 65536) {                       // W1 = L1 @ R1  (256 x 256)
    int n = i >> 8, k = i & 255;
    float acc = 0.f;
    #pragma unroll 5
    for (int r = 0; r < 70; ++r) acc += L1[n*70 + r] * R1[r*256 + k];
    wsh[WS_W1 + i] = (f16)acc;
  } else if (i < 98304) {                // W2 = L2 @ R2  (128 x 256)
    int j = i - 65536;
    int n = j >> 8, k = j & 255;
    float acc = 0.f;
    #pragma unroll
    for (int r = 0; r < 10; ++r) acc += L2[n*10 + r] * R2[r*256 + k];
    wsh[WS_W2 + j] = (f16)acc;
  } else if (i < 102400) {               // W3 (32 x 128)
    wsh[WS_W3 + (i - 98304)] = (f16)W3[i - 98304];
  } else if (i < WS_END) {               // W4 (16 x 32)
    wsh[WS_W4 + (i - 102400)] = (f16)W4[i - 102400];
  }
}

// =====================================================================
// B-fragment load helper (ISA 7.12.2: column per lane, K halves split)
// =====================================================================
__device__ __forceinline__ v16h load_bfrag(const f16* pb) {
  v16h bf;
  *(v8h*)&bf       = *(const v8h*)pb;
  *((v8h*)&bf + 1) = *(const v8h*)(pb + 16);
  return bf;
}

// =====================================================================
// One dense layer for a 16-row block of points, entirely in LDS+WMMA.
// A-fragments for the whole K range are pulled into registers first, so
// the output can overwrite the input region (LDS in-order per wave).
// - N-tiles processed in pairs: two independent WMMA accumulation
//   chains interleaved for XDL pipelining.
// - B-fragments software double-buffered: kt+1's LDS loads are issued
//   into fresh registers before kt's WMMAs, so the compiler can wait
//   with dscnt<=2 instead of draining (overlap DS with XDL).
// =====================================================================
template<int KTILES, int NTILES>
__device__ __forceinline__ void layer_pass(f16* xb, int SX,
                                           const f16* wl, int SW,
                                           const float* bias, int lane) {
  const int r     = lane & 15;
  const int koffA = (lane & 16) ? 8 : 0;    // A: 16x32 f16 layout (ISA 7.12.2)
  v16h a[KTILES];
  #pragma unroll
  for (int kt = 0; kt < KTILES; ++kt) {
    const f16* pa = xb + r*SX + kt*32 + koffA;
    *(v8h*)&a[kt]       = *(const v8h*)pa;          // K  +0..7   (ds_load_b128)
    *((v8h*)&a[kt] + 1) = *(const v8h*)(pa + 16);   // K +16..23
  }

  const int nlo    = lane & 15;
  const int koffB  = (lane & 16) ? 8 : 0;
  const int rowoff = (lane & 16) ? 8 : 0;   // D: VGPR rr -> row rr / rr+8

  if constexpr (NTILES > 1) {
    for (int nt = 0; nt < NTILES; nt += 2) {
      const f16* pbase = wl + (nt*16 + nlo)*SW + koffB;
      v8f acc0 = {}, acc1 = {};
      v16h bf0 = load_bfrag(pbase);
      v16h bf1 = load_bfrag(pbase + 16*SW);
      #pragma unroll
      for (int kt = 0; kt < KTILES; ++kt) {
        v16h nb0, nb1;
        if (kt + 1 < KTILES) {                 // prefetch next K-step's B pair
          nb0 = load_bfrag(pbase + (kt+1)*32);
          nb1 = load_bfrag(pbase + (kt+1)*32 + 16*SW);
        }
        acc0 = __builtin_amdgcn_wmma_f32_16x16x32_f16(
                   false, a[kt], false, bf0, (short)0, acc0, false, false);
        acc1 = __builtin_amdgcn_wmma_f32_16x16x32_f16(
                   false, a[kt], false, bf1, (short)0, acc1, false, false);
        if (kt + 1 < KTILES) { bf0 = nb0; bf1 = nb1; }
      }
      const int col0 = nt*16 + nlo;
      const float bs0 = bias[col0], bs1 = bias[col0 + 16];
      #pragma unroll
      for (int rr = 0; rr < 8; ++rr) {
        float v0 = fmaxf(acc0[rr] + bs0, 0.f);
        float v1 = fmaxf(acc1[rr] + bs1, 0.f);
        xb[(rr + rowoff)*SX + col0]      = (f16)v0;
        xb[(rr + rowoff)*SX + col0 + 16] = (f16)v1;
      }
    }
  } else {
    v8f acc = {};
    #pragma unroll
    for (int kt = 0; kt < KTILES; ++kt) {
      v16h bf = load_bfrag(wl + nlo*SW + kt*32 + koffB);
      acc = __builtin_amdgcn_wmma_f32_16x16x32_f16(
                false, a[kt], false, bf, (short)0, acc, false, false);
    }
    const float bs = bias[nlo];
    #pragma unroll
    for (int rr = 0; rr < 8; ++rr) {
      float v = fmaxf(acc[rr] + bs, 0.f);
      xb[(rr + rowoff)*SX + nlo] = (f16)v;
    }
  }
}

#ifdef USE_TDM
// Issue one TENSOR_LOAD_TO_LDS for a 2D f16 matrix (rows x cols, row-major,
// contiguous in global) into padded LDS rows.  pad: 4 DWORDs (8 halves)
// appended after every 2^pad_i * 8 bytes (= one row).
__device__ __forceinline__ void tdm_load_matrix(const f16* gsrc, unsigned lds_byte,
                                                unsigned rows, unsigned cols,
                                                unsigned pad_i) {
  unsigned long long ga = (unsigned long long)(__SIZE_TYPE__)gsrc;
  u32x4 g0 = { 1u,                                   // count=1, load, user D#
               lds_byte,                             // lds_addr
               (unsigned)(ga & 0xFFFFFFFFull),       // global_addr[31:0]
               (unsigned)((ga >> 32) & 0x1FFFFFFull) | (2u << 30) }; // [56:32] | type=2
  i32x8 g1 = { (int)((1u << 16) | (1u << 20) | (pad_i << 22) | (3u << 25)),
               // ^ data_size=2B, pad_enable, pad_interval, pad_amount=4 DW
               (int)((cols & 0xFFFFu) << 16),        // tensor_dim0[15:0] @ bit48
               (int)((rows & 0xFFFFu) << 16),        // dim0 hi=0 | tensor_dim1[15:0]
               (int)((cols & 0xFFFFu) << 16),        // dim1 hi=0 | tile_dim0
               (int)(rows & 0xFFFFu),                // tile_dim1 | tile_dim2=0
               (int)cols,                            // tensor_dim0_stride[31:0]
               0, 0 };
  i32x4 g2 = { 0, 0, 0, 0 };
  i32x4 g3 = { 0, 0, 0, 0 };
  i32x8 g4 = { 0, 0, 0, 0, 0, 0, 0, 0 };
  // 6-arg form (clang-23 / therock-10 headers): (g0, g1, g2, g3, g4, cpol)
  __builtin_amdgcn_tensor_load_to_lds(g0, g1, g2, g3, g4, 0);
}
#endif

// =====================================================================
// Main kernel: 8 waves/block, 16 points/wave, 128 points/block.
// =====================================================================
__launch_bounds__(256, 1)
__global__ void mlp_main(const float* __restrict__ coords,
                         const float* __restrict__ Bmat,
                         const float* __restrict__ b1, const float* __restrict__ b2,
                         const float* __restrict__ b3, const float* __restrict__ b4,
                         const float* __restrict__ W5, const float* __restrict__ b5,
                         const f16* __restrict__ wsh,
                         float* __restrict__ out) {
  extern __shared__ char smem[];
  float* F   = (float*)(smem + LDS_F);
  f16*   W1l = (f16*)(smem + LDS_W1);
  f16*   W2l = (f16*)(smem + LDS_W2);
  f16*   W3l = (f16*)(smem + LDS_W3);
  f16*   W4l = (f16*)(smem + LDS_W4);

  const int tid  = threadIdx.x;
  const int wave = tid >> 5, lane = tid & 31;

  // ---- stage f16 weights into padded LDS rows ----
#ifdef USE_TDM
  if (wave == 0) {
    tdm_load_matrix(wsh + WS_W1, LDS_W1, 256, 256, 6);  // row 512B = 2^6*8B
  } else if (wave == 1) {
    tdm_load_matrix(wsh + WS_W2, LDS_W2, 128, 256, 6);
  } else if (wave == 2) {
    tdm_load_matrix(wsh + WS_W3, LDS_W3,  32, 128, 5);  // row 256B = 2^5*8B
  } else if (wave == 3) {
    tdm_load_matrix(wsh + WS_W4, LDS_W4,  16,  32, 3);  // row  64B = 2^3*8B
  }
#else
  for (int c = tid; c < 8192; c += 256) {      // W1: 256 rows x 32 chunks
    int row = c >> 5, col8 = (c & 31) * 8;
    *(v8h*)(W1l + row*264 + col8) = *(const v8h*)(wsh + WS_W1 + row*256 + col8);
  }
  for (int c = tid; c < 4096; c += 256) {      // W2
    int row = c >> 5, col8 = (c & 31) * 8;
    *(v8h*)(W2l + row*264 + col8) = *(const v8h*)(wsh + WS_W2 + row*256 + col8);
  }
  for (int c = tid; c < 512; c += 256) {       // W3
    int row = c >> 4, col8 = (c & 15) * 8;
    *(v8h*)(W3l + row*136 + col8) = *(const v8h*)(wsh + WS_W3 + row*128 + col8);
  }
  if (tid < 64) {                              // W4
    int row = tid >> 2, col8 = (tid & 3) * 8;
    *(v8h*)(W4l + row*40 + col8) = *(const v8h*)(wsh + WS_W4 + row*32 + col8);
  }
#endif

  // ---- stage scalars / biases (f32) ----
  F[F_BMAT + tid] = Bmat[tid];                 // 256: Bmat row0 then row1
  F[F_B1   + tid] = b1[tid];                   // 256
  if (tid < 128) F[F_B2 + tid] = b2[tid];
  if (tid < 32)  F[F_B3 + tid] = b3[tid];
  if (tid < 16)  F[F_B4 + tid] = b4[tid];
  if (tid < 48)  F[F_W5 + tid] = W5[tid];
  if (tid < 3)   F[F_B5 + tid] = b5[tid];

#ifdef USE_TDM
  if (wave < 4) __builtin_amdgcn_s_wait_tensorcnt(0);
#endif
  __syncthreads();

  f16* xb = (f16*)(smem + LDS_ACT) + wave * (16 * 264);
  const int pt0 = blockIdx.x * 128 + wave * 16;

  // ---- Gaussian Fourier features: xb[i][m]=sin, xb[i][128+m]=cos ----
  for (int i = 0; i < 16; ++i) {
    const int p   = pt0 + i;
    const int b   = p >> 12;          // H*W = 4096
    const int rem = p & 4095;
    const float c0 = coords[(size_t)(b * 2)     * 4096 + rem];
    const float c1 = coords[(size_t)(b * 2 + 1) * 4096 + rem];
    #pragma unroll
    for (int mi = 0; mi < 4; ++mi) {
      const int m = lane + mi * 32;
      const float proj = TWO_PI * fmaf(c0, F[F_BMAT + m], c1 * F[F_BMAT + 128 + m]);
      xb[i*264 + m]       = (f16)__sinf(proj);
      xb[i*264 + 128 + m] = (f16)__cosf(proj);
    }
  }

  // ---- dense layers via WMMA (f16 x f16 -> f32 accumulate) ----
  layer_pass<8, 16>(xb, 264, W1l, 264, F + F_B1, lane);   // 256 -> 256
  layer_pass<8,  8>(xb, 264, W2l, 264, F + F_B2, lane);   // 256 -> 128
  layer_pass<4,  2>(xb, 264, W3l, 136, F + F_B3, lane);   // 128 -> 32
  layer_pass<1,  1>(xb, 264, W4l,  40, F + F_B4, lane);   //  32 -> 16

  // ---- final 16 -> 3 scalar layer (48 dots of length 16) ----
  #pragma unroll
  for (int it = 0; it < 2; ++it) {
    const int idx = lane + it * 32;
    if (idx < 48) {
      const int pt = idx / 3, c = idx - pt * 3;
      float acc = F[F_B5 + c];
      #pragma unroll
      for (int k = 0; k < 16; ++k)
        acc += (float)xb[pt*264 + k] * F[F_W5 + c*16 + k];
      __builtin_nontemporal_store(acc, out + (size_t)(pt0 + pt) * 3 + c);
    }
  }
}

// =====================================================================
extern "C" void kernel_launch(void* const* d_in, const int* in_sizes, int n_in,
                              void* d_out, int out_size, void* d_ws, size_t ws_size,
                              hipStream_t stream) {
  const float* coords = (const float*)d_in[0];
  const float* Bmat   = (const float*)d_in[1];
  const float* L1     = (const float*)d_in[2];
  const float* R1     = (const float*)d_in[3];
  const float* b1     = (const float*)d_in[4];
  const float* L2     = (const float*)d_in[5];
  const float* R2     = (const float*)d_in[6];
  const float* b2     = (const float*)d_in[7];
  const float* W3     = (const float*)d_in[8];
  const float* b3     = (const float*)d_in[9];
  const float* W4     = (const float*)d_in[10];
  const float* b4     = (const float*)d_in[11];
  const float* W5     = (const float*)d_in[12];
  const float* b5     = (const float*)d_in[13];
  f16*   wsh = (f16*)d_ws;
  float* out = (float*)d_out;

  (void)hipFuncSetAttribute((const void*)mlp_main,
                            hipFuncAttributeMaxDynamicSharedMemorySize, LDS_TOTAL);

  prep_weights<<<(WS_END + 255) / 256, 256, 0, stream>>>(L1, R1, L2, R2, W3, W4, wsh);
  mlp_main<<<4096, 256, LDS_TOTAL, stream>>>(coords, Bmat, b1, b2, b3, b4, W5, b5,
                                             wsh, out);
}
